// GraphEmbed_45303315038583
// MI455X (gfx1250) — compile-verified, matrix operand order checked by reading, not run on hardware
//
#include <hip/hip_runtime.h>
#include <hip/hip_bf16.h>

#define B_   64
#define NQ_  1024
#define L_   128
#define E_   8192
#define K_   2
#define LN_EPS 1e-5f

typedef __attribute__((ext_vector_type(16))) __bf16 v16bf;
typedef __attribute__((ext_vector_type(8)))  float  v8f;

// float -> bf16 (round to nearest even), stored as ushort to keep host side type-safe
static __device__ inline unsigned short f2bf(float f) {
    union { float f; unsigned u; } x; x.f = f;
    unsigned r = (x.u + 0x7FFFu + ((x.u >> 16) & 1u)) >> 16;
    return (unsigned short)r;
}

// Async global->LDS 16B copy (gfx1250 GLOBAL_LOAD_ASYNC_TO_LDS_B128, ASYNCcnt).
// ldsoff = LDS byte offset (low 32 bits of generic pointer to __shared__),
// g      = global source address.
static __device__ inline void async_ld16(unsigned ldsoff, const void* g) {
    unsigned long long ga = (unsigned long long)(size_t)g;
    asm volatile("global_load_async_to_lds_b128 %0, %1, off"
                 :: "v"(ldsoff), "v"(ga) : "memory");
}
static __device__ inline void async_wait0() {
    asm volatile("s_wait_asynccnt 0x0" ::: "memory");
}

// Build a 16x32 bf16 A-style fragment (ISA 7.12.2): lane (hi,m) holds
// K = [hi*8 .. hi*8+7] and [16+hi*8 .. 16+hi*8+7] of row m -> two 16B loads.
// B fragments for W stored N-major use the same mapping with row = N.
static __device__ inline v16bf load_frag(const unsigned short* base, int row,
                                         int stride, int kbase, int hi) {
    union { v16bf v; uint4 q[2]; } u;
    const unsigned short* p = base + row * stride + kbase + hi * 8;
    u.q[0] = *(const uint4*)(p);
    u.q[1] = *(const uint4*)(p + 16);
    return u.v;
}

#define WMMA_BF16(A_, B_frag, C_) \
    __builtin_amdgcn_wmma_f32_16x16x32_bf16(false, (A_), false, (B_frag), (short)0, (C_), false, false)

// ---------------- weight conversion (f32 -> bf16, once per call) -------------
__global__ void convert_kernel(const float* __restrict__ e_w1,
                               const float* __restrict__ n_w1,
                               const float* __restrict__ e_w2,
                               const float* __restrict__ n_w2,
                               unsigned short* __restrict__ w1m,  // (K,L,256)
                               float*          __restrict__ w1l,  // (K,L) col 256
                               unsigned short* __restrict__ nw1b, // (K,L,256)
                               unsigned short* __restrict__ ew2b, // (K,L,128)
                               unsigned short* __restrict__ nw2b) // (K,L,128)
{
    int i = blockIdx.x * blockDim.x + threadIdx.x;
    const int S0 = K_ * L_ * 257;
    const int S1 = K_ * L_ * 256;
    const int S2 = K_ * L_ * 128;
    if (i < S0) {
        int kn = i / 257, c = i % 257;
        float v = e_w1[i];
        if (c == 256) w1l[kn] = v;
        else          w1m[kn * 256 + c] = f2bf(v);
    } else if (i < S0 + S1) {
        int j = i - S0;           nw1b[j] = f2bf(n_w1[j]);
    } else if (i < S0 + S1 + S2) {
        int j = i - S0 - S1;      ew2b[j] = f2bf(e_w2[j]);
    } else if (i < S0 + S1 + 2 * S2) {
        int j = i - S0 - S1 - S2; nw2b[j] = f2bf(n_w2[j]);
    }
}

// ---------------- h init: broadcast z0 over nodes ----------------------------
__global__ void init_kernel(const float* __restrict__ z0,
                            float* __restrict__ h,
                            unsigned short* __restrict__ hbf)
{
    size_t i = (size_t)blockIdx.x * blockDim.x + threadIdx.x;
    if (i >= (size_t)B_ * NQ_ * L_) return;
    int l = (int)(i & (L_ - 1));
    int b = (int)(i / ((size_t)NQ_ * L_));
    float v = z0[b * L_ + l];
    h[i] = v;
    hbf[i] = f2bf(v);
}

// ---------------- edge MLP + scatter-add -------------------------------------
// block = 32 edges x 128 out cols, 4 waves; wave w owns cols [32w,32w+32).
// Each pair of B fragments feeds 4 WMMAs (two A row-halves x two N tiles).
__global__ __launch_bounds__(128) void edge_kernel(
    const unsigned short* __restrict__ hbf,
    const int*   __restrict__ eidx,
    const float* __restrict__ eattr,
    float*       __restrict__ magg,
    const unsigned short* __restrict__ w1,     // (L,256) bf16
    const float* __restrict__ w1last,          // (L,)  edge-attr column
    const float* __restrict__ b1,
    const unsigned short* __restrict__ w2,     // (L,128) bf16
    const float* __restrict__ b2)
{
    __shared__ __align__(16) unsigned short A[32 * 256];   // [h_src | h_tgt]
    __shared__ __align__(16) unsigned short M[32 * 128];   // relu(GEMM1) bf16
    __shared__ int   s_src[32];
    __shared__ float s_ea[32];

    int bid = blockIdx.x;
    int b  = bid >> 8;          // E_/32 = 256 tiles
    int et = bid & 255;
    int tid = threadIdx.x;
    {
        int er = tid >> 2, p = tid & 3;       // row 0..31, quarter 0..3
        int e  = et * 32 + er;
        int s  = eidx[e];
        int t  = eidx[E_ + e];
        if (p == 0) { s_src[er] = s; s_ea[er] = eattr[e]; }
        const unsigned short* hs = hbf + ((size_t)b * NQ_ + s) * L_;
        const unsigned short* ht = hbf + ((size_t)b * NQ_ + t) * L_;
        unsigned abase = (unsigned)(size_t)(&A[er * 256]);   // LDS byte offset
        #pragma unroll
        for (int c = 0; c < 4; ++c) {
            int ch = p * 4 + c;                               // 16B chunk 0..15
            async_ld16(abase + ch * 16,       hs + ch * 8);   // h_src half
            async_ld16(abase + 256 + ch * 16, ht + ch * 8);   // h_tgt half
        }
    }
    async_wait0();
    __syncthreads();

    int wave = tid >> 5, lane = tid & 31, hi = lane >> 4, nn = lane & 15;
    int col0 = wave * 32 + nn, col1 = col0 + 16;

    v8f c00 = {}, c01 = {}, c10 = {}, c11 = {};
    #pragma unroll
    for (int c = 0; c < 8; ++c) {
        int kb = c * 32;
        v16bf a0  = load_frag(A,            nn,   256, kb, hi);  // rows 0..15
        v16bf a1  = load_frag(A + 16 * 256, nn,   256, kb, hi);  // rows 16..31
        v16bf bb0 = load_frag(w1,           col0, 256, kb, hi);
        v16bf bb1 = load_frag(w1,           col1, 256, kb, hi);
        c00 = WMMA_BF16(a0, bb0, c00);
        c01 = WMMA_BF16(a0, bb1, c01);
        c10 = WMMA_BF16(a1, bb0, c10);
        c11 = WMMA_BF16(a1, bb1, c11);
    }
    // bias + edge-attr rank-1 column + relu -> LDS (bf16)
    {
        float bi0 = b1[col0], bi1 = b1[col1];
        float wl0 = w1last[col0], wl1 = w1last[col1];
        #pragma unroll
        for (int r = 0; r < 8; ++r) {
            int row = r + 8 * hi;
            float ea0 = s_ea[row], ea1 = s_ea[row + 16];
            float t00 = c00[r] + bi0 + ea0 * wl0; t00 = fmaxf(t00, 0.f);
            float t01 = c01[r] + bi1 + ea0 * wl1; t01 = fmaxf(t01, 0.f);
            float t10 = c10[r] + bi0 + ea1 * wl0; t10 = fmaxf(t10, 0.f);
            float t11 = c11[r] + bi1 + ea1 * wl1; t11 = fmaxf(t11, 0.f);
            M[row * 128 + col0]        = f2bf(t00);
            M[row * 128 + col1]        = f2bf(t01);
            M[(row + 16) * 128 + col0] = f2bf(t10);
            M[(row + 16) * 128 + col1] = f2bf(t11);
        }
    }
    __syncthreads();

    v8f d00 = {}, d01 = {}, d10 = {}, d11 = {};
    #pragma unroll
    for (int c = 0; c < 4; ++c) {
        int kb = c * 32;
        v16bf a0  = load_frag(M,            nn,   128, kb, hi);
        v16bf a1  = load_frag(M + 16 * 128, nn,   128, kb, hi);
        v16bf bb0 = load_frag(w2,           col0, 128, kb, hi);
        v16bf bb1 = load_frag(w2,           col1, 128, kb, hi);
        d00 = WMMA_BF16(a0, bb0, d00);
        d01 = WMMA_BF16(a0, bb1, d01);
        d10 = WMMA_BF16(a1, bb0, d10);
        d11 = WMMA_BF16(a1, bb1, d11);
    }
    // scatter-add m into m_agg[:, src, :]
    {
        float bb20 = b2[col0], bb21 = b2[col1];
        #pragma unroll
        for (int r = 0; r < 8; ++r) {
            int row = r + 8 * hi;
            int s0 = s_src[row], s1 = s_src[row + 16];
            float* mg0 = magg + ((size_t)b * NQ_ + s0) * L_;
            float* mg1 = magg + ((size_t)b * NQ_ + s1) * L_;
            unsafeAtomicAdd(mg0 + col0, d00[r] + bb20);
            unsafeAtomicAdd(mg0 + col1, d01[r] + bb21);
            unsafeAtomicAdd(mg1 + col0, d10[r] + bb20);
            unsafeAtomicAdd(mg1 + col1, d11[r] + bb21);
        }
    }
}

// ---------------- node MLP + layernorm + residual ----------------------------
// block = 32 nodes x 128 out cols, 4 waves
__global__ __launch_bounds__(128) void node_kernel(
    float*          __restrict__ h,
    unsigned short* __restrict__ hbf,
    const float*    __restrict__ magg,
    const unsigned short* __restrict__ w1,  // (L,256) bf16
    const float* __restrict__ b1,
    const float* __restrict__ lng,
    const float* __restrict__ lnb,
    const unsigned short* __restrict__ w2,  // (L,128) bf16
    const float* __restrict__ b2)
{
    __shared__ __align__(16) unsigned short A[32 * 256];   // [h | m_agg] bf16
    __shared__ __align__(16) float          X[32 * 128];   // pre-LN f32
    __shared__ __align__(16) unsigned short Xn[32 * 128];  // post-LN bf16
    __shared__ float s_mu[32], s_rs[32];

    int bid = blockIdx.x;
    int b  = bid >> 5;          // NQ_/32 = 32 tiles
    int nt = bid & 31;
    int tid = threadIdx.x;
    int nr = tid >> 2, p = tid & 3;          // row 0..31, quarter 0..3
    size_t rowoff = ((size_t)b * NQ_ + nt * 32 + nr) * L_;
    {
        // h half via async global->LDS
        unsigned abase = (unsigned)(size_t)(&A[nr * 256]);
        const unsigned short* hp = hbf + rowoff;
        #pragma unroll
        for (int c = 0; c < 4; ++c) {
            int ch = p * 4 + c;
            async_ld16(abase + ch * 16, hp + ch * 8);
        }
        // m_agg half: f32 -> bf16 conversion in registers
        const float* mp = magg + rowoff;
        #pragma unroll
        for (int j = 0; j < 32; ++j)
            A[nr * 256 + 128 + p * 32 + j] = f2bf(mp[p * 32 + j]);
    }
    async_wait0();
    __syncthreads();

    int wave = tid >> 5, lane = tid & 31, hi = lane >> 4, nn = lane & 15;
    int col0 = wave * 32 + nn, col1 = col0 + 16;

    v8f c00 = {}, c01 = {}, c10 = {}, c11 = {};
    #pragma unroll
    for (int c = 0; c < 8; ++c) {
        int kb = c * 32;
        v16bf a0  = load_frag(A,            nn,   256, kb, hi);
        v16bf a1  = load_frag(A + 16 * 256, nn,   256, kb, hi);
        v16bf bb0 = load_frag(w1,           col0, 256, kb, hi);
        v16bf bb1 = load_frag(w1,           col1, 256, kb, hi);
        c00 = WMMA_BF16(a0, bb0, c00);
        c01 = WMMA_BF16(a0, bb1, c01);
        c10 = WMMA_BF16(a1, bb0, c10);
        c11 = WMMA_BF16(a1, bb1, c11);
    }
    {
        float bi0 = b1[col0], bi1 = b1[col1];
        #pragma unroll
        for (int r = 0; r < 8; ++r) {
            int row = r + 8 * hi;
            float t00 = c00[r] + bi0; t00 = fmaxf(t00, 0.f);
            float t01 = c01[r] + bi1; t01 = fmaxf(t01, 0.f);
            float t10 = c10[r] + bi0; t10 = fmaxf(t10, 0.f);
            float t11 = c11[r] + bi1; t11 = fmaxf(t11, 0.f);
            X[row * 128 + col0]        = t00;
            X[row * 128 + col1]        = t01;
            X[(row + 16) * 128 + col0] = t10;
            X[(row + 16) * 128 + col1] = t11;
        }
    }
    __syncthreads();

    if (tid < 32) {   // per-row mean/var over 128 (staggered to spread banks)
        float mu = 0.f;
        #pragma unroll 4
        for (int i = 0; i < 128; ++i) mu += X[tid * 128 + ((i + tid * 4) & 127)];
        mu *= (1.f / 128.f);
        float m2 = 0.f;
        #pragma unroll 4
        for (int i = 0; i < 128; ++i) {
            float d = X[tid * 128 + ((i + tid * 4) & 127)] - mu;
            m2 += d * d;
        }
        m2 *= (1.f / 128.f);
        s_mu[tid] = mu;
        s_rs[tid] = rsqrtf(m2 + LN_EPS);
    }
    __syncthreads();
    {
        float mu = s_mu[nr], rs = s_rs[nr];
        #pragma unroll
        for (int j = 0; j < 32; ++j) {
            int col = p * 32 + j;
            float v = X[nr * 128 + col];
            v = (v - mu) * rs * lng[col] + lnb[col];
            Xn[nr * 128 + col] = f2bf(v);
        }
    }
    __syncthreads();

    v8f d00 = {}, d01 = {}, d10 = {}, d11 = {};
    #pragma unroll
    for (int c = 0; c < 4; ++c) {
        int kb = c * 32;
        v16bf a0  = load_frag(Xn,            nn,   128, kb, hi);
        v16bf a1  = load_frag(Xn + 16 * 128, nn,   128, kb, hi);
        v16bf bb0 = load_frag(w2,            col0, 128, kb, hi);
        v16bf bb1 = load_frag(w2,            col1, 128, kb, hi);
        d00 = WMMA_BF16(a0, bb0, d00);
        d01 = WMMA_BF16(a0, bb1, d01);
        d10 = WMMA_BF16(a1, bb0, d10);
        d11 = WMMA_BF16(a1, bb1, d11);
    }
    {
        float bb20 = b2[col0], bb21 = b2[col1];
        #pragma unroll
        for (int r = 0; r < 8; ++r) {
            int row = r + 8 * hi;
            size_t off0 = ((size_t)b * NQ_ + nt * 32 + row) * L_;
            size_t off1 = ((size_t)b * NQ_ + nt * 32 + row + 16) * L_;
            float v00 = d00[r] + bb20 + h[off0 + col0];
            float v01 = d01[r] + bb21 + h[off0 + col1];
            float v10 = d10[r] + bb20 + h[off1 + col0];
            float v11 = d11[r] + bb21 + h[off1 + col1];
            h[off0 + col0] = v00;  hbf[off0 + col0] = f2bf(v00);
            h[off0 + col1] = v01;  hbf[off0 + col1] = f2bf(v01);
            h[off1 + col0] = v10;  hbf[off1 + col0] = f2bf(v10);
            h[off1 + col1] = v11;  hbf[off1 + col1] = f2bf(v11);
        }
    }
}

// ---------------- final mean over nodes --------------------------------------
__global__ void mean_kernel(const float* __restrict__ h, float* __restrict__ out)
{
    int i = blockIdx.x * blockDim.x + threadIdx.x;
    if (i >= B_ * L_) return;
    int b = i / L_, l = i % L_;
    const float* pp = h + (size_t)b * NQ_ * L_ + l;
    float s = 0.f;
    for (int q = 0; q < NQ_; ++q) s += pp[(size_t)q * L_];
    out[i] = s * (1.f / NQ_);
}

extern "C" void kernel_launch(void* const* d_in, const int* in_sizes, int n_in,
                              void* d_out, int out_size, void* d_ws, size_t ws_size,
                              hipStream_t stream)
{
    const float* z0    = (const float*)d_in[0];
    const int*   eidx  = (const int*)  d_in[1];
    const float* eattr = (const float*)d_in[2];
    const float* e_w1  = (const float*)d_in[4];
    const float* e_b1  = (const float*)d_in[5];
    const float* e_w2  = (const float*)d_in[6];
    const float* e_b2  = (const float*)d_in[7];
    const float* n_w1  = (const float*)d_in[8];
    const float* n_b1  = (const float*)d_in[9];
    const float* ln_g  = (const float*)d_in[10];
    const float* ln_b  = (const float*)d_in[11];
    const float* n_w2  = (const float*)d_in[12];
    const float* n_b2  = (const float*)d_in[13];
    (void)in_sizes; (void)n_in; (void)out_size; (void)ws_size;

    char* ws = (char*)d_ws;
    size_t off = 0;
    float*          h    = (float*)(ws + off);          off += (size_t)B_ * NQ_ * L_ * 4;
    unsigned short* hbf  = (unsigned short*)(ws + off); off += (size_t)B_ * NQ_ * L_ * 2;
    float*          magg = (float*)(ws + off);          off += (size_t)B_ * NQ_ * L_ * 4;
    unsigned short* w1m  = (unsigned short*)(ws + off); off += (size_t)K_ * L_ * 256 * 2;
    float*          w1l  = (float*)(ws + off);          off += (size_t)K_ * L_ * 4;
    unsigned short* nw1b = (unsigned short*)(ws + off); off += (size_t)K_ * L_ * 256 * 2;
    unsigned short* ew2b = (unsigned short*)(ws + off); off += (size_t)K_ * L_ * 128 * 2;
    unsigned short* nw2b = (unsigned short*)(ws + off); off += (size_t)K_ * L_ * 128 * 2;

    // weights -> bf16
    {
        int total = K_ * L_ * 257 + K_ * L_ * 256 + 2 * K_ * L_ * 128;
        convert_kernel<<<(total + 255) / 256, 256, 0, stream>>>(
            e_w1, n_w1, e_w2, n_w2, w1m, w1l, nw1b, ew2b, nw2b);
    }
    // h = broadcast(z0)
    init_kernel<<<(int)(((size_t)B_ * NQ_ * L_ + 255) / 256), 256, 0, stream>>>(z0, h, hbf);

    for (int k = 0; k < K_; ++k) {
        hipMemsetAsync(magg, 0, (size_t)B_ * NQ_ * L_ * 4, stream);
        edge_kernel<<<B_ * (E_ / 32), 128, 0, stream>>>(
            hbf, eidx, eattr, magg,
            w1m + (size_t)k * L_ * 256, w1l + (size_t)k * L_, e_b1 + (size_t)k * L_,
            ew2b + (size_t)k * L_ * 128, e_b2 + (size_t)k * L_);
        node_kernel<<<B_ * (NQ_ / 32), 128, 0, stream>>>(
            h, hbf, magg,
            nw1b + (size_t)k * L_ * 256, n_b1 + (size_t)k * L_,
            ln_g + (size_t)k * L_, ln_b + (size_t)k * L_,
            nw2b + (size_t)k * L_ * 128, n_b2 + (size_t)k * L_);
    }
    mean_kernel<<<(B_ * L_ + 255) / 256, 256, 0, stream>>>(h, (float*)d_out);
}
